// T5RelativeAttention_32134945308856
// MI455X (gfx1250) — matrix operand order repaired and checked by reading, not compiled, not run on hardware
//
#include <hip/hip_runtime.h>
#include <stdint.h>

#define DIMC   768
#define NTOK   577
#define NPATCH 576
#define NPAD   608          // multiple of 32 (19 key blocks)
#define NBATCH 32
#define NHEAD  12
#define NREL   2209
#define MROWS  (NBATCH * NTOK)                           // 18464 = 577 * 32
#define QK_ELEMS   ((size_t)NBATCH * NHEAD * NPAD * 64)  // per Q/K/Vt array
#define XBF_ELEMS  ((size_t)MROWS * DIMC)                // 14,180,352
#define WQ_ELEMS   ((size_t)3 * DIMC * DIMC)             // 1,769,472
#define WP_ELEMS   ((size_t)DIMC * DIMC)                 // 589,824
#define OBUF_ELEMS ((size_t)MROWS * DIMC)
#define BIAS_ELEMS ((size_t)NHEAD * NTOK * NTOK)         // 3,995,148

typedef __attribute__((ext_vector_type(16))) __bf16 v16bf;
typedef __attribute__((ext_vector_type(8)))  float  v8f;

union FragB { v16bf v; uint32_t u[8]; uint4 q[2]; };
union FragC { v8f v; float f[8]; };

__device__ __forceinline__ uint16_t f2bf(float f) {
  uint32_t u = __float_as_uint(f);
  u += 0x7FFFu + ((u >> 16) & 1u);           // round-to-nearest-even
  return (uint16_t)(u >> 16);
}
__device__ __forceinline__ uint32_t packbf(float lo, float hi) {
  return (uint32_t)f2bf(lo) | ((uint32_t)f2bf(hi) << 16);
}

// ---------------------------------------------------------------------------
// Kernel A: bulk f32 -> bf16 conversion (8 elements / thread, 128-bit I/O)
// ---------------------------------------------------------------------------
__global__ __launch_bounds__(256) void cvt_bf16_kernel(
    const float* __restrict__ in, uint16_t* __restrict__ out, int n8)
{
  int i = blockIdx.x * blockDim.x + threadIdx.x;
  if (i >= n8) return;
  const float4* p = (const float4*)in + (size_t)i * 2;
  float4 a = p[0], b = p[1];
  uint4 o;
  o.x = packbf(a.x, a.y); o.y = packbf(a.z, a.w);
  o.z = packbf(b.x, b.y); o.w = packbf(b.z, b.w);
  ((uint4*)out)[i] = o;
}

// ---------------------------------------------------------------------------
// Kernel B: materialize the full T5 relative-bias matrix once (f32, ~16MB,
// L2-resident) so the attention kernel does coalesced loads, not gathers.
// ---------------------------------------------------------------------------
__global__ __launch_bounds__(256) void bias_build_kernel(
    const float* __restrict__ b1, const float* __restrict__ b2,
    const float* __restrict__ b3, const int* __restrict__ map,
    float* __restrict__ biasM)
{
  int idx = blockIdx.x * blockDim.x + threadIdx.x;
  if (idx >= (int)BIAS_ELEMS) return;
  int h   = idx / (NTOK * NTOK);
  int rem = idx - h * (NTOK * NTOK);
  int qi  = rem / NTOK;
  int kj  = rem - qi * NTOK;
  float v;
  if (qi == NTOK - 1)      v = b1[h * NTOK + kj];                    // cls row
  else if (kj == 0)        v = b2[h * NPATCH + qi];                  // cls col
  else                     v = b3[h * NREL + map[qi * NPATCH + (kj - 1)]];
  biasM[idx] = v;
}

// ---------------------------------------------------------------------------
// Kernel 1: qkv = x @ qkv_w^T with 2x2 register blocking (32x32 per wave).
// Pure bf16 loads -> 4 WMMAs per k-step. Scatters into Q (scaled), K, Vt.
// ---------------------------------------------------------------------------
__global__ __launch_bounds__(256) void qkv_kernel(
    const uint16_t* __restrict__ xb, const uint16_t* __restrict__ wb,
    uint16_t* __restrict__ Qo, uint16_t* __restrict__ Ko,
    uint16_t* __restrict__ Vo)
{
  const int wave = threadIdx.x >> 5;
  const int lane = threadIdx.x & 31;
  const int hf   = lane >> 4;
  const int l16  = lane & 15;
  const int m0   = blockIdx.y * 32;                       // 0..18432
  const int n0   = (blockIdx.x * 8 + wave) * 32;          // 0..2272

  FragC acc[2][2];
#pragma unroll
  for (int i = 0; i < 2; ++i)
#pragma unroll
    for (int j = 0; j < 2; ++j)
#pragma unroll
      for (int r = 0; r < 8; ++r) acc[i][j].f[r] = 0.0f;

  const uint16_t* ar0 = xb + (size_t)(m0 + l16) * DIMC;
  const uint16_t* ar1 = xb + (size_t)(m0 + 16 + l16) * DIMC;
  const uint16_t* br0 = wb + (size_t)(n0 + l16) * DIMC;
  const uint16_t* br1 = wb + (size_t)(n0 + 16 + l16) * DIMC;
  const int ao = hf * 8;
  const int bo = hf * 8;

  for (int kk = 0; kk < DIMC; kk += 32) {
    __builtin_prefetch(ar0 + kk + 512, 0, 0);
    __builtin_prefetch(br0 + kk + 512, 0, 0);
    FragB a0, a1, b0, b1;
    a0.q[0] = *(const uint4*)(ar0 + kk + ao);
    a0.q[1] = *(const uint4*)(ar0 + kk + 16 + ao);
    a1.q[0] = *(const uint4*)(ar1 + kk + ao);
    a1.q[1] = *(const uint4*)(ar1 + kk + 16 + ao);
    b0.q[0] = *(const uint4*)(br0 + kk + bo);
    b0.q[1] = *(const uint4*)(br0 + kk + bo + 16);
    b1.q[0] = *(const uint4*)(br1 + kk + bo);
    b1.q[1] = *(const uint4*)(br1 + kk + bo + 16);
    acc[0][0].v = __builtin_amdgcn_wmma_f32_16x16x32_bf16(false, a0.v, false, b0.v, (short)0, acc[0][0].v, false, false);
    acc[0][1].v = __builtin_amdgcn_wmma_f32_16x16x32_bf16(false, a0.v, false, b1.v, (short)0, acc[0][1].v, false, false);
    acc[1][0].v = __builtin_amdgcn_wmma_f32_16x16x32_bf16(false, a1.v, false, b0.v, (short)0, acc[1][0].v, false, false);
    acc[1][1].v = __builtin_amdgcn_wmma_f32_16x16x32_bf16(false, a1.v, false, b1.v, (short)0, acc[1][1].v, false, false);
  }

  const int t  = (n0 / DIMC);                 // 0=q 1=k 2=v (uniform per wave)
  const int hh = (n0 % DIMC) / 64;            // head (uniform: 32-block inside 64)
#pragma unroll
  for (int tn = 0; tn < 2; ++tn) {
    const int col = n0 + tn * 16 + l16;
    const int d   = col & 63;
#pragma unroll
    for (int tm = 0; tm < 2; ++tm) {
#pragma unroll
      for (int r = 0; r < 8; ++r) {
        const int grow = m0 + tm * 16 + r + hf * 8;
        const int b_   = grow / NTOK;
        const int n_   = grow - b_ * NTOK;
        const size_t bh = (size_t)b_ * NHEAD + hh;
        const float v = acc[tm][tn].f[r];
        if (t == 0)      Qo[(bh * NPAD + n_) * 64 + d] = f2bf(v * 0.125f);
        else if (t == 1) Ko[(bh * NPAD + n_) * 64 + d] = f2bf(v);
        else             Vo[(bh * 64 + d) * NPAD + n_] = f2bf(v);
      }
    }
  }
}

// ---------------------------------------------------------------------------
// Kernel 2: zero the padded token range (577..607) so WMMA never sees NaNs
// ---------------------------------------------------------------------------
__global__ void pad_zero_kernel(uint16_t* __restrict__ Qo,
                                uint16_t* __restrict__ Ko,
                                uint16_t* __restrict__ Vo)
{
  const int PADN  = NPAD - NTOK;                     // 31
  const int total = NBATCH * NHEAD * 64 * PADN;
  int idx = blockIdx.x * blockDim.x + threadIdx.x;
  if (idx >= total) return;
  int bh  = idx / (64 * PADN);
  int rem = idx - bh * (64 * PADN);
  int d   = rem / PADN;
  int n   = NTOK + (rem - d * PADN);
  Qo[((size_t)bh * NPAD + n) * 64 + d] = 0;
  Ko[((size_t)bh * NPAD + n) * 64 + d] = 0;
  Vo[((size_t)bh * 64 + d) * NPAD + n] = 0;
}

// ---------------------------------------------------------------------------
// Kernel 3: fused flash attention. One wave = 16 queries for one (b,h);
// online softmax over 19 key blocks of 32; P transposed through LDS.
// Bias comes from the precomputed biasM (coalesced row loads).
// ---------------------------------------------------------------------------
__global__ __launch_bounds__(128) void attn_kernel(
    const uint16_t* __restrict__ Qi, const uint16_t* __restrict__ Ki,
    const uint16_t* __restrict__ Vi, const float* __restrict__ biasM,
    uint16_t* __restrict__ Obuf)
{
  __shared__ uint16_t ldsP[4][16 * 32];              // 1KB per wave
  const int wave = threadIdx.x >> 5;
  const int lane = threadIdx.x & 31;
  const int hf   = lane >> 4;
  const int l16  = lane & 15;
  const int qt   = blockIdx.x * 4 + wave;
  if (qt >= NPAD / 16) return;                       // wave-uniform exit
  const int h = blockIdx.y;
  const int b = blockIdx.z;
  const size_t bh = (size_t)b * NHEAD + h;
  const uint16_t* Kb = Ki + bh * (size_t)NPAD * 64;
  const uint16_t* Vb = Vi + bh * (size_t)64 * NPAD;
  const float* biasH = biasM + (size_t)h * NTOK * NTOK;
  const int q0 = qt * 16;

  FragB qa0, qa1;
  {
    const uint16_t* qrow = Qi + (bh * NPAD + q0 + l16) * 64;
    const int o = hf * 8;
    qa0.q[0] = *(const uint4*)(qrow + o);
    qa0.q[1] = *(const uint4*)(qrow + 16 + o);
    qa1.q[0] = *(const uint4*)(qrow + 32 + o);
    qa1.q[1] = *(const uint4*)(qrow + 48 + o);
  }

  float mrow[8], lrow[8];
  FragC oacc[4];
#pragma unroll
  for (int r = 0; r < 8; ++r) { mrow[r] = -3.0e38f; lrow[r] = 0.0f; }
#pragma unroll
  for (int t = 0; t < 4; ++t)
#pragma unroll
    for (int r = 0; r < 8; ++r) oacc[t].f[r] = 0.0f;

  const int o2 = hf * 8;
  uint16_t* myP = ldsP[wave];

  for (int kb = 0; kb < NPAD / 32; ++kb) {
    const int base = kb * 32;
    FragC s0, s1;
    {   // S tile 0: keys base..base+15
      FragB kf0, kf1;
      const uint16_t* kr = Kb + (size_t)(base + l16) * 64;
      kf0.q[0] = *(const uint4*)(kr + o2);
      kf0.q[1] = *(const uint4*)(kr + o2 + 16);
      kf1.q[0] = *(const uint4*)(kr + 32 + o2);
      kf1.q[1] = *(const uint4*)(kr + 32 + o2 + 16);
      FragC z;
#pragma unroll
      for (int r = 0; r < 8; ++r) z.f[r] = 0.0f;
      z.v = __builtin_amdgcn_wmma_f32_16x16x32_bf16(false, qa0.v, false, kf0.v, (short)0, z.v, false, false);
      z.v = __builtin_amdgcn_wmma_f32_16x16x32_bf16(false, qa1.v, false, kf1.v, (short)0, z.v, false, false);
      s0 = z;
    }
    {   // S tile 1: keys base+16..base+31
      FragB kf0, kf1;
      const uint16_t* kr = Kb + (size_t)(base + 16 + l16) * 64;
      kf0.q[0] = *(const uint4*)(kr + o2);
      kf0.q[1] = *(const uint4*)(kr + o2 + 16);
      kf1.q[0] = *(const uint4*)(kr + 32 + o2);
      kf1.q[1] = *(const uint4*)(kr + 32 + o2 + 16);
      FragC z;
#pragma unroll
      for (int r = 0; r < 8; ++r) z.f[r] = 0.0f;
      z.v = __builtin_amdgcn_wmma_f32_16x16x32_bf16(false, qa0.v, false, kf0.v, (short)0, z.v, false, false);
      z.v = __builtin_amdgcn_wmma_f32_16x16x32_bf16(false, qa1.v, false, kf1.v, (short)0, z.v, false, false);
      s1 = z;
    }

    // add precomputed relative bias / mask (coalesced: lanes vary kj)
    const int kj0 = base + l16;
    const int kj1 = base + 16 + l16;
#pragma unroll
    for (int r = 0; r < 8; ++r) {
      const int qi = q0 + r + hf * 8;
      if (qi < NTOK) {
        const float* brow = biasH + (size_t)qi * NTOK;
        s0.f[r] = (kj0 < NTOK) ? s0.f[r] + brow[kj0] : -1.0e30f;
        s1.f[r] = (kj1 < NTOK) ? s1.f[r] + brow[kj1] : -1.0e30f;
      } else {
        s0.f[r] = -1.0e30f;
        s1.f[r] = -1.0e30f;
      }
    }

    // online softmax: a row's 16 tile-cols live in one VGPR across one
    // 16-lane half -> xor-shuffle reductions with masks 1,2,4,8.
    float sf[8];
#pragma unroll
    for (int r = 0; r < 8; ++r) {
      float mloc = fmaxf(s0.f[r], s1.f[r]);
#pragma unroll
      for (int xm = 1; xm < 16; xm <<= 1) mloc = fmaxf(mloc, __shfl_xor(mloc, xm, 32));
      const float mnew = fmaxf(mrow[r], mloc);
      const float scl  = __expf(mrow[r] - mnew);
      const float p0   = __expf(s0.f[r] - mnew);
      const float p1   = __expf(s1.f[r] - mnew);
      float ss = p0 + p1;
#pragma unroll
      for (int xm = 1; xm < 16; xm <<= 1) ss += __shfl_xor(ss, xm, 32);
      lrow[r] = lrow[r] * scl + ss;
      mrow[r] = mnew;
      sf[r]   = scl;
      s0.f[r] = p0; s1.f[r] = p1;
    }
#pragma unroll
    for (int t = 0; t < 4; ++t)
#pragma unroll
      for (int r = 0; r < 8; ++r) oacc[t].f[r] *= sf[r];

    // P: C-layout -> A-layout transpose through per-wave LDS region
#pragma unroll
    for (int r = 0; r < 8; ++r) {
      const int row = r + hf * 8;
      myP[row * 32 + l16]      = f2bf(s0.f[r]);
      myP[row * 32 + 16 + l16] = f2bf(s1.f[r]);
    }
    FragB pA;
    {
      const uint16_t* pr = myP + l16 * 32;
      const int o = hf * 8;
      pA.q[0] = *(const uint4*)(pr + o);
      pA.q[1] = *(const uint4*)(pr + 16 + o);
    }

    // O += P @ V  (Vt layout -> two 16B vector loads per B-frag)
#pragma unroll
    for (int t = 0; t < 4; ++t) {
      FragB vB;
      const uint16_t* vr = Vb + (size_t)(t * 16 + l16) * NPAD + base + o2;
      vB.q[0] = *(const uint4*)(vr);
      vB.q[1] = *(const uint4*)(vr + 16);
      oacc[t].v = __builtin_amdgcn_wmma_f32_16x16x32_bf16(false, pA.v, false, vB.v, (short)0, oacc[t].v, false, false);
    }
  }

  // normalize and store (bf16, [b,n,h*64+d] layout for the projection GEMM)
#pragma unroll
  for (int r = 0; r < 8; ++r) {
    const int n_ = q0 + r + hf * 8;
    if (n_ >= NTOK) continue;
    const float inv = 1.0f / lrow[r];
    uint16_t* orow = Obuf + ((size_t)b * NTOK + n_) * DIMC + h * 64;
    orow[0 * 16 + l16] = f2bf(oacc[0].f[r] * inv);
    orow[1 * 16 + l16] = f2bf(oacc[1].f[r] * inv);
    orow[2 * 16 + l16] = f2bf(oacc[2].f[r] * inv);
    orow[3 * 16 + l16] = f2bf(oacc[3].f[r] * inv);
  }
}

// ---------------------------------------------------------------------------
// Kernel 4: out = Obuf @ proj_w^T + proj_b, 2x2 blocking, f32 output.
// ---------------------------------------------------------------------------
__global__ __launch_bounds__(256) void proj_kernel(
    const uint16_t* __restrict__ Ob, const uint16_t* __restrict__ wb,
    const float* __restrict__ bias, float* __restrict__ out)
{
  const int wave = threadIdx.x >> 5;
  const int lane = threadIdx.x & 31;
  const int hf   = lane >> 4;
  const int l16  = lane & 15;
  const int m0   = blockIdx.y * 32;
  const int n0   = (blockIdx.x * 8 + wave) * 32;

  FragC acc[2][2];
#pragma unroll
  for (int i = 0; i < 2; ++i)
#pragma unroll
    for (int j = 0; j < 2; ++j)
#pragma unroll
      for (int r = 0; r < 8; ++r) acc[i][j].f[r] = 0.0f;

  const uint16_t* ar0 = Ob + (size_t)(m0 + l16) * DIMC;
  const uint16_t* ar1 = Ob + (size_t)(m0 + 16 + l16) * DIMC;
  const uint16_t* br0 = wb + (size_t)(n0 + l16) * DIMC;
  const uint16_t* br1 = wb + (size_t)(n0 + 16 + l16) * DIMC;
  const int ao = hf * 8;
  const int bo = hf * 8;

  for (int kk = 0; kk < DIMC; kk += 32) {
    __builtin_prefetch(ar0 + kk + 512, 0, 0);
    __builtin_prefetch(br0 + kk + 512, 0, 0);
    FragB a0, a1, b0, b1;
    a0.q[0] = *(const uint4*)(ar0 + kk + ao);
    a0.q[1] = *(const uint4*)(ar0 + kk + 16 + ao);
    a1.q[0] = *(const uint4*)(ar1 + kk + ao);
    a1.q[1] = *(const uint4*)(ar1 + kk + 16 + ao);
    b0.q[0] = *(const uint4*)(br0 + kk + bo);
    b0.q[1] = *(const uint4*)(br0 + kk + bo + 16);
    b1.q[0] = *(const uint4*)(br1 + kk + bo);
    b1.q[1] = *(const uint4*)(br1 + kk + bo + 16);
    acc[0][0].v = __builtin_amdgcn_wmma_f32_16x16x32_bf16(false, a0.v, false, b0.v, (short)0, acc[0][0].v, false, false);
    acc[0][1].v = __builtin_amdgcn_wmma_f32_16x16x32_bf16(false, a0.v, false, b1.v, (short)0, acc[0][1].v, false, false);
    acc[1][0].v = __builtin_amdgcn_wmma_f32_16x16x32_bf16(false, a1.v, false, b0.v, (short)0, acc[1][0].v, false, false);
    acc[1][1].v = __builtin_amdgcn_wmma_f32_16x16x32_bf16(false, a1.v, false, b1.v, (short)0, acc[1][1].v, false, false);
  }

#pragma unroll
  for (int tn = 0; tn < 2; ++tn) {
    const int col = n0 + tn * 16 + l16;
    const float bv = bias[col];
#pragma unroll
    for (int tm = 0; tm < 2; ++tm) {
#pragma unroll
      for (int r = 0; r < 8; ++r) {
        const int grow = m0 + tm * 16 + r + hf * 8;
        out[(size_t)grow * DIMC + col] = acc[tm][tn].f[r] + bv;
      }
    }
  }
}

// ---------------------------------------------------------------------------
extern "C" void kernel_launch(void* const* d_in, const int* in_sizes, int n_in,
                              void* d_out, int out_size, void* d_ws, size_t ws_size,
                              hipStream_t stream)
{
  (void)in_sizes; (void)n_in; (void)out_size; (void)ws_size;
  const float* x       = (const float*)d_in[0];
  const float* qkv_w   = (const float*)d_in[1];
  const float* proj_w  = (const float*)d_in[2];
  const float* proj_b  = (const float*)d_in[3];
  const float* beta1   = (const float*)d_in[4];
  const float* beta2   = (const float*)d_in[5];
  const float* beta3   = (const float*)d_in[6];
  const int*   mapping = (const int*)d_in[7];
  float* out = (float*)d_out;

  uint16_t* Xbf  = (uint16_t*)d_ws;
  uint16_t* Wq   = Xbf + XBF_ELEMS;
  uint16_t* Wp   = Wq  + WQ_ELEMS;
  uint16_t* Q    = Wp  + WP_ELEMS;
  uint16_t* K    = Q   + QK_ELEMS;
  uint16_t* Vt   = K   + QK_ELEMS;
  uint16_t* Obuf = Vt  + QK_ELEMS;
  float*    biasM = (float*)(Obuf + OBUF_ELEMS);

  // 0) precision staging: x / qkv_w / proj_w -> bf16 (done once per call)
  cvt_bf16_kernel<<<(int)((XBF_ELEMS / 8 + 255) / 256), 256, 0, stream>>>(x, Xbf, (int)(XBF_ELEMS / 8));
  cvt_bf16_kernel<<<(int)((WQ_ELEMS / 8 + 255) / 256), 256, 0, stream>>>(qkv_w, Wq, (int)(WQ_ELEMS / 8));
  cvt_bf16_kernel<<<(int)((WP_ELEMS / 8 + 255) / 256), 256, 0, stream>>>(proj_w, Wp, (int)(WP_ELEMS / 8));

  // 0b) materialize bias matrix (independent of the GEMMs)
  bias_build_kernel<<<(int)((BIAS_ELEMS + 255) / 256), 256, 0, stream>>>(
      beta1, beta2, beta3, mapping, biasM);

  // 1) QKV projection: 577 row-tiles (32) x 72 col-tiles (32), 8 waves/block
  dim3 g1(72 / 8, MROWS / 32);
  qkv_kernel<<<g1, 256, 0, stream>>>(Xbf, Wq, Q, K, Vt);

  // 2) zero padded token range
  {
    const int total = NBATCH * NHEAD * 64 * (NPAD - NTOK);
    pad_zero_kernel<<<(total + 255) / 256, 256, 0, stream>>>(Q, K, Vt);
  }

  // 3) fused attention: 38 query tiles (4 waves/block) x 12 heads x 32 batch
  dim3 g2((NPAD / 16 + 3) / 4, NHEAD, NBATCH);
  attn_kernel<<<g2, 128, 0, stream>>>(Q, K, Vt, biasM, Obuf);

  // 4) output projection: 577 x 24 tiles (32x32), 8 waves/block
  dim3 g3(24 / 8, MROWS / 32);
  proj_kernel<<<g3, 256, 0, stream>>>(Obuf, Wp, proj_b, out);
}
